// SparseLinear_85787676770758
// MI455X (gfx1250) — compile-verified, matrix operand order checked by reading, not run on hardware
//
#include <hip/hip_runtime.h>

// ---------------------------------------------------------------------------
// SparseLinear on MI455X (gfx1250):
//   y[M,N] = x[M,K] @ (W[N,K] * mask[N,K])^T + bias[N]
//   M=8192, K=4096, N=4096, fp32 in/out.
//
// bf16 pre-pass (mask folded), then V_WMMA_F32_16X16X32_BF16 GEMM:
//   block tile 128x256, 8 waves (2x4), wave tile 64x64 (16 accumulators),
//   LDS double-buffered, global->LDS via GLOBAL_LOAD_ASYNC_TO_LDS_B128
//   (ASYNCcnt) when the builtin is available.
// ---------------------------------------------------------------------------

typedef __attribute__((ext_vector_type(16))) __bf16          v16bf;
typedef __attribute__((ext_vector_type(8)))  float           v8f;
typedef __attribute__((ext_vector_type(8)))  unsigned short  ushort8;
typedef __attribute__((ext_vector_type(4)))  float           float4v;
typedef __attribute__((ext_vector_type(4)))  int             int4v;

// Exact pointer types for the async-to-LDS builtin (from the round-2 probe:
// param0 is 'int4 __device__ *' == int4 in address_space(1), non-const).
typedef __attribute__((address_space(1))) int4v* global_i4p;
typedef __attribute__((address_space(3))) int4v* lds_i4p;

static constexpr int M = 8192;
static constexpr int K = 4096;
static constexpr int N = 4096;

static constexpr int BM = 128;
static constexpr int BN = 256;
static constexpr int BK = 32;
static constexpr int NT = K / BK;       // 128 K-steps

#if __has_builtin(__builtin_amdgcn_global_load_async_to_lds_b128) && \
    __has_builtin(__builtin_amdgcn_s_wait_asynccnt)
#define USE_ASYNC_LDS 1
#else
#define USE_ASYNC_LDS 0
#endif

// ---------------- fp32 -> bf16 (round-to-nearest-even) ----------------------
static __device__ __forceinline__ unsigned short f32_to_bf16(float f) {
    union { float f; unsigned int u; } v;
    v.f = f;
    unsigned int u = v.u;
    if ((u & 0x7FFFFFFFu) > 0x7F800000u) {          // NaN: keep quiet NaN
        return (unsigned short)((u >> 16) | 0x0040u);
    }
    u += 0x7FFFu + ((u >> 16) & 1u);                // RNE
    return (unsigned short)(u >> 16);
}

// ---------------- pre-pass: x -> bf16 ---------------------------------------
__global__ __launch_bounds__(256)
void cvt_x_kernel(const float* __restrict__ x, unsigned short* __restrict__ xb,
                  int nvec4) {
    int i      = blockIdx.x * blockDim.x + threadIdx.x;
    int stride = gridDim.x * blockDim.x;
    for (int j = i; j < nvec4; j += stride) {
        float4v v = ((const float4v*)x)[j];
        unsigned short r0 = f32_to_bf16(v.x);
        unsigned short r1 = f32_to_bf16(v.y);
        unsigned short r2 = f32_to_bf16(v.z);
        unsigned short r3 = f32_to_bf16(v.w);
        ((unsigned long long*)xb)[j] =
            (unsigned long long)r0 | ((unsigned long long)r1 << 16) |
            ((unsigned long long)r2 << 32) | ((unsigned long long)r3 << 48);
    }
}

// ---------------- pre-pass: W * mask -> bf16 --------------------------------
__global__ __launch_bounds__(256)
void cvt_w_kernel(const float* __restrict__ w, const int* __restrict__ mask,
                  unsigned short* __restrict__ wb, int nvec4) {
    int i      = blockIdx.x * blockDim.x + threadIdx.x;
    int stride = gridDim.x * blockDim.x;
    for (int j = i; j < nvec4; j += stride) {
        float4v v = ((const float4v*)w)[j];
        int4v   m = ((const int4v*)mask)[j];
        unsigned short r0 = m.x ? f32_to_bf16(v.x) : (unsigned short)0;
        unsigned short r1 = m.y ? f32_to_bf16(v.y) : (unsigned short)0;
        unsigned short r2 = m.z ? f32_to_bf16(v.z) : (unsigned short)0;
        unsigned short r3 = m.w ? f32_to_bf16(v.w) : (unsigned short)0;
        ((unsigned long long*)wb)[j] =
            (unsigned long long)r0 | ((unsigned long long)r1 << 16) |
            ((unsigned long long)r2 << 32) | ((unsigned long long)r3 << 48);
    }
}

// Build a 16x32 bf16 A/B fragment for one lane from a K-major LDS row.
// Lane l (row = l%16, lh = l/16) holds K chunks [lh*8, lh*8+8) and
// [lh*8+16, lh*8+24): two 16-byte ds_load_b128's.
static __device__ __forceinline__ v16bf load_frag(const unsigned short* p) {
    union { ushort8 h[2]; v16bf v; } u;
    u.h[0] = *(const ushort8*)(p);
    u.h[1] = *(const ushort8*)(p + 16);
    return u.v;
}

// ---------------- WMMA GEMM -------------------------------------------------
// grid = (N/BN, M/BM) = (16, 64), block = 256 threads = 8 waves (2Mx4N).
// Each wave computes a 64x64 slab: 4x4 grid of 16x16 f32 accumulators.
__global__ __launch_bounds__(256)
void sparse_linear_wmma_kernel(const unsigned short* __restrict__ xb,
                               const unsigned short* __restrict__ wb,
                               const float* __restrict__ bias,
                               float* __restrict__ out) {
    __shared__ alignas(16) unsigned short sA[2][BM][BK];   //  8 KB / buffer
    __shared__ alignas(16) unsigned short sB[2][BN][BK];   // 16 KB / buffer

    const int tid   = threadIdx.x;
    const int lane  = tid & 31;
    const int wave  = tid >> 5;
    const int waveM = wave >> 2;        // 0..1
    const int waveN = wave & 3;         // 0..3
    const int lm    = lane & 15;
    const int lh    = lane >> 4;

    const int bm = blockIdx.y * BM;
    const int bn = blockIdx.x * BN;

    // Staging map: 16B chunks of a (rows x 32) bf16 tile; 4 chunks per row.
    // A tile: 512 chunks -> 2/thread. B tile: 1024 chunks -> 4/thread.
    int rowA[2], colA[2], rowB[4], colB[4];
#pragma unroll
    for (int i = 0; i < 2; ++i) {
        const int c = tid + i * 256;
        rowA[i] = c >> 2;  colA[i] = (c & 3) * 8;
    }
#pragma unroll
    for (int j = 0; j < 4; ++j) {
        const int c = tid + j * 256;
        rowB[j] = c >> 2;  colB[j] = (c & 3) * 8;
    }

    // Accumulators preloaded with bias (all 8 elements of a C VGPR share N).
    v8f acc[4][4];
#pragma unroll
    for (int n = 0; n < 4; ++n) {
        const float bv = bias[bn + waveN * 64 + n * 16 + lm];
#pragma unroll
        for (int m = 0; m < 4; ++m) {
            acc[m][n] = (v8f){bv, bv, bv, bv, bv, bv, bv, bv};
        }
    }

    // One K-step of MACs out of LDS buffer b: 16 ds_load_b128 -> 16 WMMAs.
    auto compute_step = [&](int b) {
        v16bf bfrag[4];
#pragma unroll
        for (int n = 0; n < 4; ++n) {
            bfrag[n] = load_frag(&sB[b][waveN * 64 + n * 16 + lm][lh * 8]);
        }
#pragma unroll
        for (int m = 0; m < 4; ++m) {
            const v16bf afrag =
                load_frag(&sA[b][waveM * 64 + m * 16 + lm][lh * 8]);
#pragma unroll
            for (int n = 0; n < 4; ++n) {
                acc[m][n] = __builtin_amdgcn_wmma_f32_16x16x32_bf16(
                    /*neg_a=*/false, afrag,
                    /*neg_b=*/false, bfrag[n],
                    /*c_mod=*/(short)0, acc[m][n],
                    /*reuse_a=*/false, /*reuse_b=*/false);
            }
        }
    };

#if USE_ASYNC_LDS
    // gfx1250 path: GLOBAL_LOAD_ASYNC_TO_LDS_B128 writes LDS directly,
    // tracked by ASYNCcnt -> no VGPR staging, no loadcnt stall in the loop.
    auto issue_tile = [&](int kt, int b) {
        const size_t k0 = (size_t)kt * BK;
#pragma unroll
        for (int i = 0; i < 2; ++i) {
            __builtin_amdgcn_global_load_async_to_lds_b128(
                (global_i4p)(xb + (size_t)(bm + rowA[i]) * K + k0 + colA[i]),
                (lds_i4p)&sA[b][rowA[i]][colA[i]],
                0, 0);
        }
#pragma unroll
        for (int j = 0; j < 4; ++j) {
            __builtin_amdgcn_global_load_async_to_lds_b128(
                (global_i4p)(wb + (size_t)(bn + rowB[j]) * K + k0 + colB[j]),
                (lds_i4p)&sB[b][rowB[j]][colB[j]],
                0, 0);
        }
    };

    issue_tile(0, 0);
    int buf = 0;
    for (int kt = 0; kt < NT; ++kt) {
        __builtin_amdgcn_s_wait_asynccnt(0);   // our tile kt writes landed
        __syncthreads();                       // everyone's tile kt visible
        if (kt + 1 < NT) issue_tile(kt + 1, buf ^ 1);  // overlaps compute
        compute_step(buf);
        __syncthreads();   // readers of buf done before it is refilled
        buf ^= 1;
    }
#else
    // Fallback: global->VGPR->LDS with register prefetch, double buffer,
    // two barriers per step (writes to buf at step kt are ordered after
    // step kt-1's barrier; last readers of buf ran at step kt-2).
    ushort8 ra[2], rb[4];
    auto fetch_tile = [&](int kt) {
        const size_t k0 = (size_t)kt * BK;
#pragma unroll
        for (int i = 0; i < 2; ++i)
            ra[i] = *(const ushort8*)(xb + (size_t)(bm + rowA[i]) * K + k0 + colA[i]);
#pragma unroll
        for (int j = 0; j < 4; ++j)
            rb[j] = *(const ushort8*)(wb + (size_t)(bn + rowB[j]) * K + k0 + colB[j]);
    };

    fetch_tile(0);
    int buf = 0;
    for (int kt = 0; kt < NT; ++kt) {
#pragma unroll
        for (int i = 0; i < 2; ++i)
            *(ushort8*)&sA[buf][rowA[i]][colA[i]] = ra[i];
#pragma unroll
        for (int j = 0; j < 4; ++j)
            *(ushort8*)&sB[buf][rowB[j]][colB[j]] = rb[j];
        __syncthreads();
        if (kt + 1 < NT) fetch_tile(kt + 1);
        compute_step(buf);
        __syncthreads();
        buf ^= 1;
    }
#endif

    // Epilogue: C VGPR i holds (M = base + lh*8 + i, N = base + lm).
#pragma unroll
    for (int m = 0; m < 4; ++m) {
#pragma unroll
        for (int n = 0; n < 4; ++n) {
            const int col     = bn + waveN * 64 + n * 16 + lm;
            const int rowbase = bm + waveM * 64 + m * 16 + lh * 8;
#pragma unroll
            for (int i = 0; i < 8; ++i) {
                out[(size_t)(rowbase + i) * N + col] = acc[m][n][i];
            }
        }
    }
}

// ---------------------------------------------------------------------------
extern "C" void kernel_launch(void* const* d_in, const int* in_sizes, int n_in,
                              void* d_out, int out_size, void* d_ws, size_t ws_size,
                              hipStream_t stream) {
    const float* x    = (const float*)d_in[0];   // [M,K]
    const float* w    = (const float*)d_in[1];   // [N,K]
    const float* bias = (const float*)d_in[2];   // [N]
    const int*   mask = (const int*)d_in[3];     // [N,K]
    float*       out  = (float*)d_out;           // [M,N]

    // Workspace: bf16 x (64 MB) then bf16 masked W (32 MB, L2-resident).
    unsigned short* xb = (unsigned short*)d_ws;
    unsigned short* wb = (unsigned short*)((char*)d_ws + (size_t)M * K * sizeof(unsigned short));

    {
        const int nvec4 = (M * K) / 4;
        cvt_x_kernel<<<4096, 256, 0, stream>>>(x, xb, nvec4);
    }
    {
        const int nvec4 = (N * K) / 4;
        cvt_w_kernel<<<4096, 256, 0, stream>>>(w, mask, wb, nvec4);
    }

    dim3 grid(N / BN, M / BM);                   // (16, 64)
    sparse_linear_wmma_kernel<<<grid, 256, 0, stream>>>(xb, wb, bias, out);
}
